// WindowAttention_82849919139849
// MI455X (gfx1250) — matrix operand order repaired and checked by reading, not compiled, not run on hardware
//
#include <hip/hip_runtime.h>

#define DEVI __device__ __forceinline__

typedef __attribute__((ext_vector_type(16))) _Float16 v16h;
typedef __attribute__((ext_vector_type(8)))  _Float16 v8h;
typedef __attribute__((ext_vector_type(8)))  float    v8f;

#define WS_    7
#define SHIFT_ 3
#define RES_   56
#define DIMS_  192
#define HEADS_ 6
#define HD_    32
#define NTOK   49          // real tokens
#define NPAD   64          // padded tokens
#define WMAT   (DIMS_*DIMS_)       // 36864 elements per weight matrix
#define WT_BYTES   (4 * WMAT * 2)  // 294912 B of f16 weights in ws
#define BIAS_ELEMS (HEADS_ * NPAD * NPAD)  // 24576 f32

union V16U { v16h v; v8h h[2]; };

// A-fragment (16x32 f16): lane<16 holds row M=lane, K {k0..k0+7, k0+16..k0+23};
// lane>=16 holds row M=lane-16, K {k0+8..k0+15, k0+24..k0+31}. Row-major source.
DEVI v16h load_a_frag(const _Float16* base, int stride, int row0, int k0, int lane) {
    int half = lane >> 4;
    const _Float16* p = base + (row0 + (lane & 15)) * stride + k0 + half * 8;
    V16U u;
    u.h[0] = *(const v8h*)(p);
    u.h[1] = *(const v8h*)(p + 16);
    return u.v;
}

// B-fragment (32x16 f16): lane holds column n = n0 + (lane&15);
// lanes 0-15 carry K {k0..k0+7, k0+16..k0+23}, lanes 16-31 carry K {k0+8..k0+15, k0+24..k0+31}.
// Source laid out src[n][k] contiguous in k (stride = row pitch).
DEVI v16h load_b_frag(const _Float16* base, int stride, int n0, int k0, int lane) {
    const _Float16* p = base + (n0 + (lane & 15)) * stride + k0 + (lane >> 4) * 8;
    V16U u;
    u.h[0] = *(const v8h*)(p);
    u.h[1] = *(const v8h*)(p + 16);
    return u.v;
}

DEVI v8f wmma16(v16h a, v16h b, v8f c) {
    return __builtin_amdgcn_wmma_f32_16x16x32_f16(false, a, false, b, (short)0, c,
                                                  false, false);
}

DEVI int shift_region(int y) {           // region id along one axis (shifted coords)
    return (y < RES_ - WS_) ? 0 : ((y < RES_ - SHIFT_) ? 1 : 2);
}

// ---------------------------------------------------------------------------
// Kernel 1a: convert weights fp32 (K x N row-major) -> f16 transposed Wt[n][k]
// ---------------------------------------------------------------------------
__global__ void swin_convert_weights(const float* __restrict__ wq,
                                     const float* __restrict__ wk,
                                     const float* __restrict__ wv,
                                     const float* __restrict__ wo,
                                     _Float16* __restrict__ out) {
    int i = blockIdx.x * 256 + threadIdx.x;
    if (i >= 4 * WMAT) return;
    int mm  = i / WMAT;
    int rem = i - mm * WMAT;
    int n   = rem / DIMS_;
    int k   = rem - n * DIMS_;
    const float* src = (mm == 0) ? wq : (mm == 1) ? wk : (mm == 2) ? wv : wo;
    out[mm * WMAT + n * DIMS_ + k] = (_Float16)src[k * DIMS_ + n];
}

// ---------------------------------------------------------------------------
// Kernel 1b: padded rel-pos bias table bias_pad[h][m(64)][n(64)],
//            pad rows/cols pre-filled with -1e30 (acts as -inf after softmax)
// ---------------------------------------------------------------------------
__global__ void swin_build_bias(const float* __restrict__ table,
                                float* __restrict__ bias_pad) {
    int i = blockIdx.x * 256 + threadIdx.x;
    if (i >= BIAS_ELEMS) return;
    int h   = i / (NPAD * NPAD);
    int rem = i - h * (NPAD * NPAD);
    int m   = rem / NPAD;
    int n   = rem - m * NPAD;
    float v = -1.0e30f;
    if (m < NTOK && n < NTOK) {
        int rm = m / WS_, cm = m - rm * WS_;
        int rn = n / WS_, cn = n - rn * WS_;
        int idx = (rm - rn + WS_ - 1) * (2 * WS_ - 1) + (cm - cn + WS_ - 1);
        v = table[idx * HEADS_ + h];
    }
    bias_pad[i] = v;
}

// ---------------------------------------------------------------------------
// Kernel 2: one block per window (4096 blocks x 256 threads = 8 waves)
// ---------------------------------------------------------------------------
__global__ __launch_bounds__(256)
void swin_window_attention(const float* __restrict__ x,
                           const float* __restrict__ bq,
                           const float* __restrict__ bk,
                           const float* __restrict__ bv,
                           const float* __restrict__ bo,
                           const _Float16* __restrict__ wt,     // 4 transposed f16 mats
                           const float* __restrict__ bias_pad,  // [6][64][64]
                           float* __restrict__ out) {
    __shared__ _Float16 s_xo[NPAD * DIMS_];        // X, later aliased as O (f16)
    __shared__ _Float16 s_q [NPAD * DIMS_];        // Q row-major, 1/sqrt(hd) folded
    __shared__ _Float16 s_k [NPAD * DIMS_];        // K row-major
    __shared__ _Float16 s_vt[DIMS_ * NPAD];        // V transposed: [dim][token]
    __shared__ _Float16 s_p [HEADS_ * NPAD * NPAD]; // softmax probs per head
    __shared__ float s_bq[DIMS_], s_bk[DIMS_], s_bv[DIMS_], s_bo[DIMS_];
    __shared__ int   s_cnt[NPAD];                  // shift-mask region code per token

    const int tid   = threadIdx.x;
    const int lane  = tid & 31;
    const int wave  = tid >> 5;
    const int laneN = lane & 15;
    const int half  = lane >> 4;

    const int w    = blockIdx.x;     // window id 0..4095
    const int bimg = w >> 6;         // image
    const int win  = w & 63;
    const int wy   = win >> 3;
    const int wx   = win & 7;

    // ---- Phase 1: stage biases, region codes, and shifted-window X ----
    if (tid < DIMS_) {
        s_bq[tid] = bq[tid];
        s_bk[tid] = bk[tid];
        s_bv[tid] = bv[tid];
        s_bo[tid] = bo[tid];
    }
    if (tid < NPAD) {
        int cnt = 0;
        if (tid < NTOK) {
            int r = tid / WS_, c = tid - r * WS_;
            cnt = shift_region(wy * WS_ + r) * 3 + shift_region(wx * WS_ + c);
        }
        s_cnt[tid] = cnt;
    }
    for (int i = tid; i < NPAD * DIMS_; i += 256) {
        int row = i / DIMS_;
        int col = i - row * DIMS_;
        float v = 0.0f;
        if (row < NTOK) {
            int r = row / WS_, c = row - r * WS_;
            int sy = wy * WS_ + r + SHIFT_; if (sy >= RES_) sy -= RES_;
            int sx = wx * WS_ + c + SHIFT_; if (sx >= RES_) sx -= RES_;
            v = x[((bimg * RES_ + sy) * RES_ + sx) * DIMS_ + col];
        }
        s_xo[i] = (_Float16)v;
    }
    __syncthreads();

    // ---- Phase 2: QKV projections (144 16x16 output tiles over 8 waves) ----
    const float qscale = 0.17677669529663687f;     // 1/sqrt(32)
    for (int job = wave; job < 144; job += 8) {
        int mat = job / 48;            // 0=Q 1=K 2=V
        int rem = job - mat * 48;
        int mi  = rem / 12;
        int ni  = rem - mi * 12;
        const _Float16* wmat = wt + mat * WMAT;
        v8f acc = {};
#pragma unroll
        for (int ks = 0; ks < 6; ++ks) {
            v16h a = load_a_frag(s_xo, DIMS_, mi * 16, ks * 32, lane);
            v16h b = load_b_frag(wmat, DIMS_, ni * 16, ks * 32, lane);
            acc = wmma16(a, b, acc);
        }
        int col = ni * 16 + laneN;
        if (mat == 0) {
            float bias = s_bq[col];
#pragma unroll
            for (int j = 0; j < 8; ++j) {
                int m = mi * 16 + j + half * 8;
                s_q[m * DIMS_ + col] = (_Float16)((acc[j] + bias) * qscale);
            }
        } else if (mat == 1) {
            float bias = s_bk[col];
#pragma unroll
            for (int j = 0; j < 8; ++j) {
                int m = mi * 16 + j + half * 8;
                s_k[m * DIMS_ + col] = (_Float16)(acc[j] + bias);
            }
        } else {
            float bias = s_bv[col];
            v8h pk;
#pragma unroll
            for (int j = 0; j < 8; ++j) pk[j] = (_Float16)(acc[j] + bias);
            // tokens contiguous in j -> one 16B LDS store into V^T
            *(v8h*)(&s_vt[col * NPAD + mi * 16 + half * 8]) = pk;
        }
    }
    __syncthreads();

    // ---- Phase 3: scores + softmax, one wave per head ----
    if (wave < HEADS_) {
        const int h = wave;
        const float* bp_h = bias_pad + h * (NPAD * NPAD);
        _Float16* ph = s_p + h * NPAD * NPAD;
#pragma unroll
        for (int mi = 0; mi < 4; ++mi) {
            v8f sc[4];
#pragma unroll
            for (int ni = 0; ni < 4; ++ni) {
                // C preload: precomputed padded bias + cheap region mask
                int n   = ni * 16 + laneN;
                int cn_ = s_cnt[n];
                const float* bp = bp_h + (mi * 16 + half * 8) * NPAD + n;
                v8f c;
#pragma unroll
                for (int j = 0; j < 8; ++j) {
                    float bm = bp[j * NPAD];
                    if (s_cnt[mi * 16 + j + half * 8] != cn_) bm -= 1000.0f;
                    c[j] = bm;
                }
                v16h a = load_a_frag(s_q, DIMS_, mi * 16, h * HD_, lane);
                v16h b = load_b_frag(s_k, DIMS_, ni * 16, h * HD_, lane);
                sc[ni] = wmma16(a, b, c);
            }
            // row softmax: in-lane over 4 N-tiles + xor-shuffle over 16 lanes
#pragma unroll
            for (int j = 0; j < 8; ++j) {
                float mx = sc[0][j];
#pragma unroll
                for (int ni = 1; ni < 4; ++ni) mx = fmaxf(mx, sc[ni][j]);
#pragma unroll
                for (int d = 8; d >= 1; d >>= 1) mx = fmaxf(mx, __shfl_xor(mx, d, 32));
                float sum = 0.0f;
#pragma unroll
                for (int ni = 0; ni < 4; ++ni) {
                    float e = __expf(sc[ni][j] - mx);
                    sc[ni][j] = e;
                    sum += e;
                }
#pragma unroll
                for (int d = 8; d >= 1; d >>= 1) sum += __shfl_xor(sum, d, 32);
                float rs = 1.0f / sum;
#pragma unroll
                for (int ni = 0; ni < 4; ++ni) sc[ni][j] *= rs;
            }
            // write probs (f16) row-major for the A-fragment of P@V
#pragma unroll
            for (int ni = 0; ni < 4; ++ni) {
                int n = ni * 16 + laneN;
#pragma unroll
                for (int j = 0; j < 8; ++j) {
                    int m = mi * 16 + j + half * 8;
                    ph[m * NPAD + n] = (_Float16)sc[ni][j];
                }
            }
        }
    }
    __syncthreads();

    // ---- Phase 4: O_h = P @ V_h, write O (f16) into s_xo (X is dead) ----
    if (wave < HEADS_) {
        const int h = wave;
        const _Float16* ph = s_p + h * NPAD * NPAD;
#pragma unroll
        for (int mi = 0; mi < 4; ++mi) {
#pragma unroll
            for (int ni2 = 0; ni2 < 2; ++ni2) {
                v8f acc = {};
#pragma unroll
                for (int ki = 0; ki < 2; ++ki) {
                    v16h a = load_a_frag(ph, NPAD, mi * 16, ki * 32, lane);
                    v16h b = load_b_frag(s_vt, NPAD, h * HD_ + ni2 * 16, ki * 32, lane);
                    acc = wmma16(a, b, acc);
                }
                int col = h * HD_ + ni2 * 16 + laneN;
#pragma unroll
                for (int j = 0; j < 8; ++j) {
                    int m = mi * 16 + j + half * 8;
                    s_xo[m * DIMS_ + col] = (_Float16)acc[j];
                }
            }
        }
    }
    __syncthreads();

    // ---- Phase 5: output projection + store with reverse shift ----
    const _Float16* wmo = wt + 3 * WMAT;
    for (int job = wave; job < 48; job += 8) {
        int mi = job / 12;
        int ni = job - mi * 12;
        v8f acc = {};
#pragma unroll
        for (int ks = 0; ks < 6; ++ks) {
            v16h a = load_a_frag(s_xo, DIMS_, mi * 16, ks * 32, lane);
            v16h b = load_b_frag(wmo, DIMS_, ni * 16, ks * 32, lane);
            acc = wmma16(a, b, acc);
        }
        int col  = ni * 16 + laneN;
        float bias = s_bo[col];
#pragma unroll
        for (int j = 0; j < 8; ++j) {
            int m = mi * 16 + j + half * 8;
            if (m < NTOK) {
                int r = m / WS_, c = m - (m / WS_) * WS_;
                int sy = wy * WS_ + r + SHIFT_; if (sy >= RES_) sy -= RES_;
                int sx = wx * WS_ + c + SHIFT_; if (sx >= RES_) sx -= RES_;
                out[((bimg * RES_ + sy) * RES_ + sx) * DIMS_ + col] = acc[j] + bias;
            }
        }
    }
}

// ---------------------------------------------------------------------------
extern "C" void kernel_launch(void* const* d_in, const int* in_sizes, int n_in,
                              void* d_out, int out_size, void* d_ws, size_t ws_size,
                              hipStream_t stream) {
    const float* x     = (const float*)d_in[0];
    const float* table = (const float*)d_in[1];
    const float* wq    = (const float*)d_in[2];
    const float* bq    = (const float*)d_in[3];
    const float* wk    = (const float*)d_in[4];
    const float* bk    = (const float*)d_in[5];
    const float* wv    = (const float*)d_in[6];
    const float* bv    = (const float*)d_in[7];
    const float* wo    = (const float*)d_in[8];
    const float* bo    = (const float*)d_in[9];
    float* out = (float*)d_out;

    _Float16* wt    = (_Float16*)d_ws;                       // 288 KB
    float* bias_pad = (float*)((char*)d_ws + WT_BYTES);      // +96 KB

    int nconv = 4 * WMAT;
    swin_convert_weights<<<(nconv + 255) / 256, 256, 0, stream>>>(wq, wk, wv, wo, wt);
    swin_build_bias<<<(BIAS_ELEMS + 255) / 256, 256, 0, stream>>>(table, bias_pad);

    swin_window_attention<<<4096, 256, 0, stream>>>(x, bq, bk, bv, bo, wt, bias_pad, out);

    (void)in_sizes; (void)n_in; (void)out_size; (void)ws_size;
}
